// newLayer_2147483648123
// MI455X (gfx1250) — matrix operand order repaired and checked by reading, not compiled, not run on hardware
//
#include <hip/hip_runtime.h>
#include <stdint.h>

// ---------------------------------------------------------------------------
// out = x @ (bf16(q) )^T * scale,   M=8192 N=4096 K=4096
// f32 accuracy recovered via x = x_hi(bf16) + x_lo(bf16) split:
//   acc = wmma(x_hi, W) ; acc = wmma(x_lo, W)   (f32 accumulate)
// ---------------------------------------------------------------------------

typedef __attribute__((ext_vector_type(16))) __bf16 bf16x16;
typedef __attribute__((ext_vector_type(8)))  __bf16 bf16x8;
typedef __attribute__((ext_vector_type(4)))  __bf16 bf16x4;
typedef __attribute__((ext_vector_type(8)))  float  f32x8;
typedef __attribute__((ext_vector_type(4)))  float  f32x4;
typedef __attribute__((ext_vector_type(4)))  int    i32x4;

static constexpr int Mdim = 8192;
static constexpr int Ndim = 4096;
static constexpr int Kdim = 4096;
static constexpr int BM = 128, BN = 128, BK = 32;
static constexpr int LDK = BK + 8;        // padded LDS row stride (40 bf16 = 80B)
static constexpr int NKT = Kdim / BK;     // 128 k-tiles

union Frag { bf16x16 v; bf16x8 h[2]; };

// -------------------- prepass: dequantize W int32 -> bf16 (exact, codes<127)
__global__ __launch_bounds__(256) void convW_kernel(const int* __restrict__ q,
                                                    __bf16* __restrict__ w) {
  size_t i = ((size_t)blockIdx.x * 256 + threadIdx.x) * 4;
  i32x4 v = *(const i32x4*)(q + i);
  bf16x4 b;
  b[0] = (__bf16)(float)v[0];
  b[1] = (__bf16)(float)v[1];
  b[2] = (__bf16)(float)v[2];
  b[3] = (__bf16)(float)v[3];
  *(bf16x4*)(w + i) = b;
}

// -------------------- main GEMM ---------------------------------------------
template <bool PRECONV_W>
__global__ __launch_bounds__(256) void gemm_kernel(
    const float* __restrict__ x, const int* __restrict__ q,
    const __bf16* __restrict__ wbf, const float* __restrict__ scale,
    float* __restrict__ out) {

  __shared__ __bf16 Ahi[2][BM * LDK];
  __shared__ __bf16 Alo[2][BM * LDK];
  __shared__ __bf16 Bsh[2][BN * LDK];

  const int t    = threadIdx.x;
  const int lane = t & 31;
  const int wave = t >> 5;
  const int wm0  = (wave & 3) * 32;   // 4 waves along M
  const int wn0  = (wave >> 2) * 64;  // 2 waves along N
  const int bm0  = blockIdx.y * BM;
  const int bn0  = blockIdx.x * BN;

  const float s = scale[0];

  // global->LDS staging maps (256 threads cover a 128x32 tile)
  const int ar  = t >> 3;          // row base (0..31), +32 per rep
  const int akq = (t & 7) * 4;     // k offset in elements (0..28 step 4)
  const int bcol  = t >> 2;        // (async path) col base (0..63), +64 per rep
  const int bkq8  = (t & 3) * 8;   // (async path) k offset (0,8,16,24)

  f32x4 areg[4];
  i32x4 breg[4];

  auto loadA = [&](int kt) {
    const float* p = x + (size_t)(bm0 + ar) * Kdim + (size_t)kt * BK + akq;
#pragma unroll
    for (int i = 0; i < 4; ++i)
      areg[i] = *(const f32x4*)(p + (size_t)32 * i * Kdim);
  };
  auto storeA = [&](int buf) {
#pragma unroll
    for (int i = 0; i < 4; ++i) {
      bf16x4 h, l;
#pragma unroll
      for (int j = 0; j < 4; ++j) {
        float  f  = areg[i][j];
        __bf16 hb = (__bf16)f;
        h[j] = hb;
        l[j] = (__bf16)(f - (float)hb);
      }
      const int off = (ar + 32 * i) * LDK + akq;
      *(bf16x4*)&Ahi[buf][off] = h;
      *(bf16x4*)&Alo[buf][off] = l;
    }
  };
  auto loadB = [&](int kt) {   // fallback: int32 weights from global
    const int* p = q + (size_t)(bn0 + ar) * Kdim + (size_t)kt * BK + akq;
#pragma unroll
    for (int i = 0; i < 4; ++i)
      breg[i] = *(const i32x4*)(p + (size_t)32 * i * Kdim);
  };
  auto storeB = [&](int buf) {
#pragma unroll
    for (int i = 0; i < 4; ++i) {
      bf16x4 b;
#pragma unroll
      for (int j = 0; j < 4; ++j) b[j] = (__bf16)(float)breg[i][j];
      *(bf16x4*)&Bsh[buf][(ar + 32 * i) * LDK + akq] = b;
    }
  };
  auto asyncB = [&](int kt, int buf) {  // pre-dequantized bf16 W: DMA to LDS
#pragma unroll
    for (int i = 0; i < 2; ++i) {
      const int col = bcol + 64 * i;
      const __bf16* gp = wbf + (size_t)(bn0 + col) * Kdim + (size_t)kt * BK + bkq8;
      unsigned ldsa = (unsigned)(uintptr_t)&Bsh[buf][col * LDK + bkq8];
      uint64_t ga   = (uint64_t)(uintptr_t)gp;
      asm volatile("global_load_async_to_lds_b128 %0, %1, off"
                   :: "v"(ldsa), "v"(ga) : "memory");
    }
  };

  f32x8 acc[2][4] = {};

  auto compute = [&](int buf) {
    Frag ah[2], al[2], bfr[4];
    const int rA = lane & 15;
    const int kA = (lane < 16) ? 0 : 8;    // A: K {g..g+7, g+16..g+23}
#pragma unroll
    for (int mt = 0; mt < 2; ++mt) {
      const __bf16* ph = &Ahi[buf][(wm0 + mt * 16 + rA) * LDK + kA];
      ah[mt].h[0] = *(const bf16x8*)ph;
      ah[mt].h[1] = *(const bf16x8*)(ph + 16);
      const __bf16* pl = &Alo[buf][(wm0 + mt * 16 + rA) * LDK + kA];
      al[mt].h[0] = *(const bf16x8*)pl;
      al[mt].h[1] = *(const bf16x8*)(pl + 16);
    }
    const int kB = (lane < 16) ? 0 : 8;    // B: K {g..g+7, g+16..g+23}
#pragma unroll
    for (int nt = 0; nt < 4; ++nt) {
      const __bf16* pb = &Bsh[buf][(wn0 + nt * 16 + rA) * LDK + kB];
      bfr[nt].h[0] = *(const bf16x8*)pb;
      bfr[nt].h[1] = *(const bf16x8*)(pb + 16);
    }
#pragma unroll
    for (int mt = 0; mt < 2; ++mt)
#pragma unroll
      for (int nt = 0; nt < 4; ++nt) {
        acc[mt][nt] = __builtin_amdgcn_wmma_f32_16x16x32_bf16(
            false, ah[mt].v, false, bfr[nt].v, (short)0, acc[mt][nt], false, false);
        acc[mt][nt] = __builtin_amdgcn_wmma_f32_16x16x32_bf16(
            false, al[mt].v, false, bfr[nt].v, (short)0, acc[mt][nt], false, false);
      }
  };

  // ---- prologue: stage k-tile 0 into buffer 0
  loadA(0);
  if constexpr (PRECONV_W) asyncB(0, 0); else loadB(0);
  storeA(0);
  if constexpr (!PRECONV_W) storeB(0);
  if constexpr (PRECONV_W) asm volatile("s_wait_asynccnt 0x0" ::: "memory");
  __syncthreads();

  // ---- double-buffered main loop
  for (int kt = 0; kt < NKT; ++kt) {
    const int  cur  = kt & 1;
    const int  nxt  = cur ^ 1;
    const bool more = (kt + 1) < NKT;
    if (more) {
      loadA(kt + 1);
      if constexpr (PRECONV_W) asyncB(kt + 1, nxt);
      else loadB(kt + 1);
    }
    compute(cur);
    if (more) {
      storeA(nxt);
      if constexpr (!PRECONV_W) storeB(nxt);
      if constexpr (PRECONV_W) asm volatile("s_wait_asynccnt 0x0" ::: "memory");
    }
    __syncthreads();
  }

  // ---- epilogue: C layout -> VGPR r: lanes0-15 M=r, lanes16-31 M=r+8; N=lane&15
  const int cn    = lane & 15;
  const int mhalf = (lane >> 4) * 8;
#pragma unroll
  for (int mt = 0; mt < 2; ++mt)
#pragma unroll
    for (int nt = 0; nt < 4; ++nt) {
      const int    gn = bn0 + wn0 + nt * 16 + cn;
      const size_t gm = (size_t)(bm0 + wm0 + mt * 16 + mhalf);
#pragma unroll
      for (int r = 0; r < 8; ++r)
        out[(gm + r) * Ndim + gn] = acc[mt][nt][r] * s;
    }
}

// -------------------- host launcher -----------------------------------------
extern "C" void kernel_launch(void* const* d_in, const int* in_sizes, int n_in,
                              void* d_out, int out_size, void* d_ws, size_t ws_size,
                              hipStream_t stream) {
  const float* x     = (const float*)d_in[0];
  const int*   q     = (const int*)d_in[1];
  const float* scale = (const float*)d_in[2];
  float*       out   = (float*)d_out;

  const size_t needW = (size_t)Ndim * Kdim * sizeof(__bf16);  // 32 MB
  dim3 grid(Ndim / BN, Mdim / BM);  // 32 x 64
  dim3 block(256);

  if (ws_size >= needW) {
    __bf16* wbf = (__bf16*)d_ws;
    const int nconv = (Ndim * Kdim) / 4 / 256;  // 16384 blocks
    convW_kernel<<<nconv, 256, 0, stream>>>(q, wbf);
    gemm_kernel<true><<<grid, block, 0, stream>>>(x, q, wbf, scale, out);
  } else {
    gemm_kernel<false><<<grid, block, 0, stream>>>(x, q, nullptr, scale, out);
  }
}
